// WindowAttention_4449586118908
// MI455X (gfx1250) — matrix operand order repaired and checked by reading, not compiled, not run on hardware
//
#include <hip/hip_runtime.h>

// ---------------------------------------------------------------------------
// Window attention (Swin-style), MI455X / gfx1250, wave32 + WMMA f16->f32.
//   0) one-time f32->f16 conversion of w_qkv / w_proj
//   1) QKV GEMM: LDS-staged A-strip + double-buffered ASYNC-to-LDS B tiles
//   2) per-(b,h,branch) attention: WMMA S, f32 softmax, LDS-staged P, WMMA P@V
//   3) output projection: same pipelined WMMA GEMM
// ---------------------------------------------------------------------------

typedef __attribute__((ext_vector_type(16))) _Float16 v16h;
typedef __attribute__((ext_vector_type(8)))  float    v8f;

constexpr int    kB  = 2048;
constexpr int    kN  = 49;
constexpr int    kC  = 384;
constexpr int    kH  = 12;
constexpr int    kD  = 32;          // head dim
constexpr int    kNW = 64;
constexpr int    k3C = 3 * kC;      // 1152
constexpr int    kLP = kC + 8;      // padded LDS row stride (halves)
constexpr float  kScale = 0.17677669529663687f;               // 32^-0.5
constexpr size_t kHeadElems = (size_t)kB * kH * kN * kD;      // per q/k/v tensor (f16)
constexpr size_t kXElems    = (size_t)kB * kN * kC;           // fp32 x output elems
constexpr int    kWQKV16    = kC * k3C;                       // 442368 halves
constexpr int    kWPROJ16   = kC * kC;                        // 147456 halves

#if defined(__gfx1250__) && __has_builtin(__builtin_amdgcn_global_load_async_to_lds_b32)
#define HAVE_ASYNC_LDS 1
#else
#define HAVE_ASYNC_LDS 0
#endif

union Frag {
  v16h          v;
  _Float16      h[16];
  unsigned int  u[8];
};

// 16-bit A-operand K index for half j (0..15) in lane-half kh (lane>>4).
__device__ __forceinline__ int a_kmap(int j, int kh) {
  return kh * 16 + j;
}

__device__ __forceinline__ unsigned int pack2h(float x, float y) {
  union { _Float16 h[2]; unsigned int u; } p;
  p.h[0] = (_Float16)x; p.h[1] = (_Float16)y;
  return p.u;
}

// 4-byte memory -> LDS copy: CDNA5 async path (ASYNCcnt) with sync fallback.
__device__ __forceinline__ void copy_b32_to_lds(const void* g, void* l) {
#if HAVE_ASYNC_LDS
  __builtin_amdgcn_global_load_async_to_lds_b32(
      (__attribute__((address_space(1))) int*)g,
      (__attribute__((address_space(3))) int*)l, 0, 0);
#else
  *(unsigned int*)l = *(const unsigned int*)g;
#endif
}

__device__ __forceinline__ void wait_async_copies() {
#if HAVE_ASYNC_LDS
 #if __has_builtin(__builtin_amdgcn_s_wait_asynccnt)
  __builtin_amdgcn_s_wait_asynccnt(0);
 #else
  asm volatile("s_wait_asynccnt 0" ::: "memory");
 #endif
#endif
}

// Stage one 384xK-row x 16-col f16 weight tile (natural row-major, 32 B/row)
// into LDS: 12288 B = 24 per-thread B32 copies for a 128-thread block.
template <int ROWBYTES>
__device__ __forceinline__ void stage_b_tile(const _Float16* wbase, int ntile,
                                             _Float16* ldsdst, int tid) {
  const char* g = (const char*)wbase + (size_t)ntile * 32;
  char*       l = (char*)ldsdst;
#pragma unroll
  for (int it = 0; it < 24; ++it) {
    const int fi  = it * 512 + tid * 4;   // flat byte index in 12288-B tile
    const int row = fi >> 5;              // K row
    const int col = fi & 31;              // byte within 16-half row
    copy_b32_to_lds(g + (size_t)row * ROWBYTES + col, l + fi);
  }
}

// ---------------------------------------------------------------------------
// Kernel 0: one-time weight conversion to f16 (weights are GEMM-reused ~6k x).
// ---------------------------------------------------------------------------
__global__ __launch_bounds__(256) void w16_convert_kernel(
    const float* __restrict__ w_qkv, const float* __restrict__ w_proj,
    _Float16* __restrict__ w16qkv, _Float16* __restrict__ w16proj)
{
  const int i = blockIdx.x * 256 + threadIdx.x;
  if (i < kWQKV16)  w16qkv[i]  = (_Float16)w_qkv[i];
  if (i < kWPROJ16) w16proj[i] = (_Float16)w_proj[i];
}

// ---------------------------------------------------------------------------
// Kernel 1: Y = X_t @ w_qkv + b_qkv, scattered into f16 Q(scaled)/K/V [B,H,N,D]
// grid = (1568, 2); block = 128 (4 waves). Block owns a 64-row M-strip staged
// once in LDS (f16); B tiles stream in via double-buffered async-to-LDS.
// ---------------------------------------------------------------------------
__global__ __launch_bounds__(128) void qkv_wmma_kernel(
    const float* __restrict__ x1, const float* __restrict__ x2,
    const _Float16* __restrict__ w16, const float* __restrict__ b_qkv,
    _Float16* __restrict__ Qs,
    _Float16* __restrict__ K1, _Float16* __restrict__ V1,
    _Float16* __restrict__ K2, _Float16* __restrict__ V2)
{
  __shared__ _Float16 Alds[64][kLP];        // 64 x 384 f16 A-strip  (50.2 KB)
  __shared__ _Float16 Bnat[2][kC][16];      // double-buffered B tile (24 KB)

  const int t     = blockIdx.y;
  const int m0blk = blockIdx.x * 64;
  const float* __restrict__ X = t ? x2 : x1;
  const int tid = threadIdx.x;

  // ---- stage A strip: 2 threads per row, f32 -> packed f16 ---------------
  {
    const int row = tid >> 1;
    const int cb  = (tid & 1) * (kC / 2);
    const float* __restrict__ src = X + (size_t)(m0blk + row) * kC + cb;
    for (int cc = 0; cc < kC / 2; cc += 2)
      *(unsigned int*)&Alds[row][cb + cc] = pack2h(src[cc], src[cc + 1]);
  }

  const int wave = tid >> 5;
  const int lane = tid & 31;
  const int nl = lane & 15, kh = lane >> 4;

  const int ntStart = t ? 24 : 0;             // q of x2 is never used
  int cur = 0;
  stage_b_tile<k3C * 2>(w16, ntStart, &Bnat[cur][0][0], tid);

  for (int nt = ntStart; nt < 72; ++nt) {
    wait_async_copies();
    __syncthreads();                          // A + Bnat[cur] visible, other free
    if (nt + 1 < 72) {
      stage_b_tile<k3C * 2>(w16, nt + 1, &Bnat[cur ^ 1][0][0], tid);
      __builtin_prefetch(w16 + (size_t)(nt + 1) * 16, 0, 3);
    }

    const _Float16* __restrict__ Bc = &Bnat[cur][0][0];
    v8f acc = {0.f, 0.f, 0.f, 0.f, 0.f, 0.f, 0.f, 0.f};
#pragma unroll
    for (int kk = 0; kk < kC; kk += 32) {
      Frag a, b;
#pragma unroll
      for (int g = 0; g < 8; ++g)
        a.u[g] = *(const unsigned int*)&Alds[wave * 16 + nl][kk + a_kmap(2 * g, kh)];
#pragma unroll
      for (int j = 0; j < 16; ++j)
        b.h[j] = Bc[(kk + kh * 16 + j) * 16 + nl];
      acc = __builtin_amdgcn_wmma_f32_16x16x32_f16(false, a.v, false, b.v,
                                                   (short)0, acc, false, false);
    }

    // ---- scatter epilogue ------------------------------------------------
    const int   c      = nt * 16 + nl;
    const int   qkvSel = nt / 24;             // 0=q 1=k 2=v
    const int   hh     = (c % kC) / kD;
    const int   d      = c % kD;
    const float bias   = b_qkv[c];
    _Float16* __restrict__ dst;
    float scale = 1.f;
    if (qkvSel == 0)      { dst = Qs; scale = kScale; }
    else if (qkvSel == 1) { dst = t ? K2 : K1; }
    else                  { dst = t ? V2 : V1; }
#pragma unroll
    for (int r = 0; r < 8; ++r) {
      const int mm = m0blk + wave * 16 + r + 8 * kh;   // < 100352 always
      const int bi = mm / kN, ns = mm % kN;
      dst[(((size_t)bi * kH + hh) * kN + ns) * kD + d] =
          (_Float16)((acc[r] + bias) * scale);
    }
    cur ^= 1;
  }
}

// ---------------------------------------------------------------------------
// Kernel 2: per-(b,h,t) attention. 4 waves, wave w owns query rows 16w..16w+15.
// grid = (2048, 12, 2); block = 128.
// ---------------------------------------------------------------------------
__global__ __launch_bounds__(128) void attn_wmma_kernel(
    const _Float16* __restrict__ Qs,
    const _Float16* __restrict__ K1, const _Float16* __restrict__ V1,
    const _Float16* __restrict__ K2, const _Float16* __restrict__ V2,
    const float* __restrict__ mask1, const float* __restrict__ mask2,
    const float* __restrict__ rpb_table, const int* __restrict__ rel_index,
    float* __restrict__ weights_out)
{
  __shared__ _Float16 Plds[64][72];   // 64x64 probs, padded rows

  const int b = blockIdx.x, h = blockIdx.y, t = blockIdx.z;
  const int w    = threadIdx.x >> 5;
  const int lane = threadIdx.x & 31;
  const int nl = lane & 15, kh = lane >> 4;

  const _Float16* __restrict__ Kx = t ? K2 : K1;
  const _Float16* __restrict__ Vx = t ? V2 : V1;
  const float* __restrict__ mask =
      (t ? mask2 : mask1) + (size_t)(b % kNW) * kN * kN;
  const size_t headOff = (((size_t)b * kH + h) * kN) * kD;
  const _Float16* __restrict__ Qb = Qs + headOff;
  const _Float16* __restrict__ Kb = Kx + headOff;
  const _Float16* __restrict__ Vb = Vx + headOff;

  // ---- A operand: 16 query rows (f16, already scaled) ---------------------
  Frag qa;
  {
    const int  m     = w * 16 + nl;
    const bool valid = m < kN;
    const _Float16* qrow = Qb + (size_t)m * kD;
#pragma unroll
    for (int g = 0; g < 8; ++g) {
      const int k = a_kmap(2 * g, kh);
      qa.u[g] = valid ? *(const unsigned int*)(qrow + k) : 0u;
    }
  }

  // ---- S tiles + bias/mask ------------------------------------------------
  float sv[4][8];
#pragma unroll
  for (int ct = 0; ct < 4; ++ct) {
    const int  key    = ct * 16 + nl;
    const bool kvalid = key < kN;
    const _Float16* krow = Kb + (size_t)key * kD + kh * 16;
    Frag kb;
#pragma unroll
    for (int g = 0; g < 8; ++g)
      kb.u[g] = kvalid ? *(const unsigned int*)(krow + 2 * g) : 0u;
    v8f z = {0.f, 0.f, 0.f, 0.f, 0.f, 0.f, 0.f, 0.f};
    const v8f s = __builtin_amdgcn_wmma_f32_16x16x32_f16(false, qa.v, false, kb.v,
                                                         (short)0, z, false, false);
#pragma unroll
    for (int r = 0; r < 8; ++r) {
      const int me = w * 16 + r + 8 * kh;     // query row of this D element
      float val = -1e30f;
      if (kvalid && me < kN) {
        const int ridx = rel_index[me * kN + key];
        val = s[r] + rpb_table[ridx * kH + h] + mask[me * kN + key];
      }
      sv[ct][r] = val;
    }
  }

  // ---- row softmax: a row's 49 scores live in one half-wave (16 lanes x 4)
#pragma unroll
  for (int r = 0; r < 8; ++r) {
    float mx = fmaxf(fmaxf(sv[0][r], sv[1][r]), fmaxf(sv[2][r], sv[3][r]));
#pragma unroll
    for (int off = 8; off >= 1; off >>= 1)
      mx = fmaxf(mx, __shfl_xor(mx, off, 16));
    float p[4], sum = 0.f;
#pragma unroll
    for (int ct = 0; ct < 4; ++ct) { p[ct] = __expf(sv[ct][r] - mx); sum += p[ct]; }
#pragma unroll
    for (int off = 8; off >= 1; off >>= 1)
      sum += __shfl_xor(sum, off, 16);
    const float inv = 1.f / sum;
    const int   me  = w * 16 + r + 8 * kh;
#pragma unroll
    for (int ct = 0; ct < 4; ++ct)
      Plds[me][ct * 16 + nl] = (_Float16)(p[ct] * inv);
  }
  __syncthreads();

  // ---- O = P @ V ----------------------------------------------------------
  v8f oacc0 = {0.f, 0.f, 0.f, 0.f, 0.f, 0.f, 0.f, 0.f};
  v8f oacc1 = {0.f, 0.f, 0.f, 0.f, 0.f, 0.f, 0.f, 0.f};
#pragma unroll
  for (int s = 0; s < 2; ++s) {               // K-steps: keys 0-31, 32-63
    Frag pa;
    const int ma = w * 16 + nl;
#pragma unroll
    for (int g = 0; g < 8; ++g) {
      const int j = 2 * g;
      const int k = s * 32 + a_kmap(j, kh);
      pa.h[j]     = Plds[ma][k];
      pa.h[j + 1] = Plds[ma][k + 1];
    }
#pragma unroll
    for (int ctv = 0; ctv < 2; ++ctv) {       // output cols d 0-15 / 16-31
      Frag vb;
      const int d = ctv * 16 + nl;
#pragma unroll
      for (int g = 0; g < 8; ++g) {
        const int j   = 2 * g;
        const int key = s * 32 + kh * 16 + j;
        vb.h[j]     = (key     < kN) ? Vb[(size_t)key * kD + d]       : (_Float16)0.f;
        vb.h[j + 1] = (key + 1 < kN) ? Vb[(size_t)(key + 1) * kD + d] : (_Float16)0.f;
      }
      v8f& oa = ctv ? oacc1 : oacc0;
      oa = __builtin_amdgcn_wmma_f32_16x16x32_f16(false, pa.v, false, vb.v,
                                                  (short)0, oa, false, false);
    }
  }

  // ---- store fp32 `weights` [B, N, 2C] ------------------------------------
#pragma unroll
  for (int ctv = 0; ctv < 2; ++ctv) {
    const v8f oa = ctv ? oacc1 : oacc0;
    const int d  = ctv * 16 + nl;
#pragma unroll
    for (int r = 0; r < 8; ++r) {
      const int me = w * 16 + r + 8 * kh;
      if (me < kN)
        weights_out[((size_t)b * kN + me) * (2 * kC) + t * kC + h * kD + d] = oa[r];
    }
  }
}

// ---------------------------------------------------------------------------
// Kernel 3: x = weights[:, :, :C] @ w_proj + b_proj   (fp32 out)
// grid = (1568); block = 128. Same pipelined structure as kernel 1.
// ---------------------------------------------------------------------------
__global__ __launch_bounds__(128) void proj_wmma_kernel(
    const float* __restrict__ weights, const _Float16* __restrict__ w16p,
    const float* __restrict__ b_proj, float* __restrict__ xout)
{
  __shared__ _Float16 Alds[64][kLP];
  __shared__ _Float16 Bnat[2][kC][16];

  const int m0blk = blockIdx.x * 64;
  const int tid = threadIdx.x;

  {
    const int row = tid >> 1;
    const int cb  = (tid & 1) * (kC / 2);
    const float* __restrict__ src = weights + (size_t)(m0blk + row) * (2 * kC) + cb;
    for (int cc = 0; cc < kC / 2; cc += 2)
      *(unsigned int*)&Alds[row][cb + cc] = pack2h(src[cc], src[cc + 1]);
  }

  const int wave = tid >> 5;
  const int lane = tid & 31;
  const int nl = lane & 15, kh = lane >> 4;

  int cur = 0;
  stage_b_tile<kC * 2>(w16p, 0, &Bnat[cur][0][0], tid);

  for (int nt = 0; nt < 24; ++nt) {
    wait_async_copies();
    __syncthreads();
    if (nt + 1 < 24) {
      stage_b_tile<kC * 2>(w16p, nt + 1, &Bnat[cur ^ 1][0][0], tid);
      __builtin_prefetch(w16p + (size_t)(nt + 1) * 16, 0, 3);
    }

    const _Float16* __restrict__ Bc = &Bnat[cur][0][0];
    v8f acc = {0.f, 0.f, 0.f, 0.f, 0.f, 0.f, 0.f, 0.f};
#pragma unroll
    for (int kk = 0; kk < kC; kk += 32) {
      Frag a, b;
#pragma unroll
      for (int g = 0; g < 8; ++g)
        a.u[g] = *(const unsigned int*)&Alds[wave * 16 + nl][kk + a_kmap(2 * g, kh)];
#pragma unroll
      for (int j = 0; j < 16; ++j)
        b.h[j] = Bc[(kk + kh * 16 + j) * 16 + nl];
      acc = __builtin_amdgcn_wmma_f32_16x16x32_f16(false, a.v, false, b.v,
                                                   (short)0, acc, false, false);
    }

    const int   c    = nt * 16 + nl;
    const float bias = b_proj[c];
#pragma unroll
    for (int r = 0; r < 8; ++r) {
      const int mm = m0blk + wave * 16 + r + 8 * kh;
      xout[(size_t)mm * kC + c] = acc[r] + bias;
    }
    cur ^= 1;
  }
}

// ---------------------------------------------------------------------------
extern "C" void kernel_launch(void* const* d_in, const int* in_sizes, int n_in,
                              void* d_out, int out_size, void* d_ws, size_t ws_size,
                              hipStream_t stream) {
  (void)in_sizes; (void)n_in; (void)out_size; (void)ws_size;
  const float* x1        = (const float*)d_in[0];
  const float* x2        = (const float*)d_in[1];
  const float* mask1     = (const float*)d_in[2];
  const float* mask2     = (const float*)d_in[3];
  const float* w_qkv     = (const float*)d_in[4];
  const float* b_qkv     = (const float*)d_in[5];
  const float* w_proj    = (const float*)d_in[6];
  const float* b_proj    = (const float*)d_in[7];
  const float* rpb_table = (const float*)d_in[8];
  const int*   rel_index = (const int*)d_in[9];

  _Float16* ws  = (_Float16*)d_ws;
  _Float16* Qs  = ws;
  _Float16* K1  = ws + 1 * kHeadElems;
  _Float16* V1  = ws + 2 * kHeadElems;
  _Float16* K2  = ws + 3 * kHeadElems;
  _Float16* V2  = ws + 4 * kHeadElems;
  _Float16* w16qkv  = ws + 5 * kHeadElems;
  _Float16* w16proj = w16qkv + kWQKV16;

  float* xout = (float*)d_out;             // [2048, 49, 384]
  float* wout = xout + kXElems;            // [2048, 49, 768]

  // 0) one-time f16 conversion of the GEMM weights
  w16_convert_kernel<<<dim3((kWQKV16 + 255) / 256), 256, 0, stream>>>(
      w_qkv, w_proj, w16qkv, w16proj);

  // 1) QKV projection for both inputs (pipelined WMMA GEMM)
  qkv_wmma_kernel<<<dim3((kB * kN) / 64, 2), 128, 0, stream>>>(
      x1, x2, w16qkv, b_qkv, Qs, K1, V1, K2, V2);

  // 2) Attention per (batch, head, branch)
  attn_wmma_kernel<<<dim3(kB, kH, 2), 128, 0, stream>>>(
      Qs, K1, V1, K2, V2, mask1, mask2, rpb_table, rel_index, wout);

  // 3) Output projection from the first half of `weights`
  proj_wmma_kernel<<<dim3((kB * kN) / 64), 128, 0, stream>>>(
      wout, w16proj, b_proj, xout);
}
